// ProtoDynGNN_1443109012197
// MI455X (gfx1250) — matrix operand order repaired and checked by reading, not compiled, not run on hardware
//
#include <hip/hip_runtime.h>
#include <hip/hip_bf16.h>

typedef __attribute__((ext_vector_type(16))) _Float16 v16h;
typedef __attribute__((ext_vector_type(8)))  float    v8f;

#define PI_F 3.14159f
#define TWO_PI_F (2.0f * 3.14159f)

// ---------------------------------------------------------------------------
// WMMA fragment helpers (CDNA5 wave32 layouts, 16x16x32 f16 -> f32)
// A 16x32 (MxK): lanes 0-15 rows M=lane, K = kb..kb+7 / kb+16..kb+23 (kb=0)
//                lanes 16-31 rows M=lane-16, kb=8
// B 32x16 (KxN): lanes 0-15 N=lane, K = kb..kb+7 / kb+16..kb+23 (kb=0); lanes 16-31 kb=8
// C/D 16x16:     lane -> N=lane&15; vgpr r -> M = r + 8*(lane>=16)
// ---------------------------------------------------------------------------
__device__ __forceinline__ v16h load_a_frag(const _Float16* ldsA, int lda,
                                            int lane, int k0) {
  int m  = lane & 15;
  int kb = k0 + ((lane & 16) ? 8 : 0);
  const _Float16* r = ldsA + m * lda + kb;
  v16h a;
#pragma unroll
  for (int t = 0; t < 8; ++t) a[t] = r[t];
#pragma unroll
  for (int t = 0; t < 8; ++t) a[8 + t] = r[16 + t];
  return a;
}

// B pre-packed fragment-major: Bp[((ntile*kt + kk)*32 + lane)*16 + t]
__device__ __forceinline__ v16h load_b_packed(const _Float16* Bp, int kt,
                                              int ntile, int kk, int lane) {
  const v16h* p =
      (const v16h*)(Bp + ((((size_t)ntile * kt + kk) * 32 + lane) << 4));
  return *p;
}

__device__ __forceinline__ float wrapf(float a) {
  float t = fmodf(a + PI_F, TWO_PI_F);
  if (t < 0.0f) t += TWO_PI_F;
  return t - PI_F;
}

// ---------------------------------------------------------------------------
// Weight packing: B[K][128] f32 -> fragment-ordered f16
// ---------------------------------------------------------------------------
__global__ void pack_b_kernel(const float* __restrict__ B,
                              _Float16* __restrict__ Bp, int K) {
  int i = blockIdx.x * blockDim.x + threadIdx.x;
  if (i >= 128 * K) return;
  int t = i & 15;
  int lane = (i >> 4) & 31;
  int rest = i >> 9;  // ntile*(K/32) + kk
  int kt = K >> 5;
  int kk = rest % kt;
  int ntile = rest / kt;
  int n = ntile * 16 + (lane & 15);
  int kb = kk * 32 + ((lane & 16) ? 8 : 0) + ((t < 8) ? t : 16 + (t - 8));
  Bp[i] = (_Float16)B[kb * 128 + n];
}

__global__ void fill_kernel(float* __restrict__ p, float v, int n) {
  int i = blockIdx.x * blockDim.x + threadIdx.x;
  if (i < n) p[i] = v;
}

// ---------------------------------------------------------------------------
// Node featurization: hat_h = LN(cat(V_com,SBF_chi)) @ sc_W + sc_b   [N,128]
//                     h     = LN(cat(Xca,Vcb,SBFphi,SBFpsi)) @ bb_W + bb_b
// ---------------------------------------------------------------------------
__global__ void node_init_kernel(
    const float* __restrict__ Vcom, const float* __restrict__ SBFchi,
    const float* __restrict__ Xca, const float* __restrict__ Vcb,
    const float* __restrict__ SBFphi, const float* __restrict__ SBFpsi,
    const float* __restrict__ scg, const float* __restrict__ scbeta,
    const float* __restrict__ scW, const float* __restrict__ scb,
    const float* __restrict__ bbg, const float* __restrict__ bbbeta,
    const float* __restrict__ bbW, const float* __restrict__ bbb,
    float* __restrict__ hat_h, float* __restrict__ h, int N) {
  __shared__ float s_sc[19];
  __shared__ float s_bb[38];
  int n = blockIdx.x;
  int t = threadIdx.x;
  if (n >= N) return;
  if (t < 19) s_sc[t] = (t < 3) ? Vcom[n * 3 + t] : SBFchi[n * 16 + (t - 3)];
  if (t < 38) {
    float v;
    if (t < 3)       v = Xca[n * 3 + t];
    else if (t < 6)  v = Vcb[n * 3 + (t - 3)];
    else if (t < 22) v = SBFphi[n * 16 + (t - 6)];
    else             v = SBFpsi[n * 16 + (t - 22)];
    s_bb[t] = v;
  }
  __syncthreads();
  float m1 = 0.f;
  for (int k = 0; k < 19; ++k) m1 += s_sc[k];
  m1 *= (1.0f / 19.0f);
  float v1 = 0.f;
  for (int k = 0; k < 19; ++k) { float d = s_sc[k] - m1; v1 += d * d; }
  v1 *= (1.0f / 19.0f);
  float rs1 = rsqrtf(v1 + 1e-5f);
  float acc1 = scb[t];
  for (int k = 0; k < 19; ++k)
    acc1 += ((s_sc[k] - m1) * rs1 * scg[k] + scbeta[k]) * scW[k * 128 + t];
  hat_h[n * 128 + t] = acc1;

  float m2 = 0.f;
  for (int k = 0; k < 38; ++k) m2 += s_bb[k];
  m2 *= (1.0f / 38.0f);
  float v2 = 0.f;
  for (int k = 0; k < 38; ++k) { float d = s_bb[k] - m2; v2 += d * d; }
  v2 *= (1.0f / 38.0f);
  float rs2 = rsqrtf(v2 + 1e-5f);
  float acc2 = bbb[t];
  for (int k = 0; k < 38; ++k)
    acc2 += ((s_bb[k] - m2) * rs2 * bbg[k] + bbbeta[k]) * bbW[k * 128 + t];
  h[n * 128 + t] = acc2;
}

// ---------------------------------------------------------------------------
// Edge MLP: hat_e = relu(LN(e_in) @ eW1 + eb1) @ eW2 + eb2   [E,128]
// 16 edges/block, 256 threads. Layer-1 scalar (K=66), layer-2 WMMA.
// ---------------------------------------------------------------------------
__global__ void edge_mlp_kernel(
    const float* __restrict__ d_sc, const float* __restrict__ d_min,
    const float* __restrict__ res_table,
    const int* __restrict__ ri, const int* __restrict__ rj,
    const float* __restrict__ elng, const float* __restrict__ elnb,
    const float* __restrict__ eW1, const float* __restrict__ eb1,
    const _Float16* __restrict__ eW2p, const float* __restrict__ eb2,
    float* __restrict__ hat_e, int E) {
  __shared__ float raw[16][66];
  __shared__ float nrm[16][66];
  __shared__ _Float16 A[16][128];
  __shared__ float s_mean[16], s_rstd[16];
  int e0 = blockIdx.x * 16;
  int tid = threadIdx.x;
  for (int i = tid; i < 16 * 66; i += 256) {
    int r = i / 66, f = i % 66;
    int ge = e0 + r;
    float v = 0.f;
    if (ge < E) {
      if (f == 0)      v = d_sc[ge];
      else if (f == 1) v = d_min[ge];
      else if (f < 34) v = res_table[ri[ge] * 32 + (f - 2)];
      else             v = res_table[rj[ge] * 32 + (f - 34)];
    }
    raw[r][f] = v;
  }
  __syncthreads();
  if (tid < 16) {
    float m = 0.f;
    for (int k = 0; k < 66; ++k) m += raw[tid][k];
    m *= (1.0f / 66.0f);
    float vv = 0.f;
    for (int k = 0; k < 66; ++k) { float d = raw[tid][k] - m; vv += d * d; }
    vv *= (1.0f / 66.0f);
    s_mean[tid] = m;
    s_rstd[tid] = rsqrtf(vv + 1e-5f);
  }
  __syncthreads();
  for (int i = tid; i < 16 * 66; i += 256) {
    int r = i / 66, f = i % 66;
    nrm[r][f] = (raw[r][f] - s_mean[r]) * s_rstd[r] * elng[f] + elnb[f];
  }
  __syncthreads();
  {
    int r = tid >> 4, cb = tid & 15;
#pragma unroll
    for (int j = 0; j < 8; ++j) {
      int col = cb + 16 * j;
      float acc = eb1[col];
      for (int k = 0; k < 66; ++k) acc += nrm[r][k] * eW1[k * 128 + col];
      A[r][col] = (_Float16)fmaxf(acc, 0.0f);
    }
  }
  __syncthreads();
  int lane = tid & 31, wv = tid >> 5;
  int n0 = wv * 16;
  v8f c = {};
#pragma unroll
  for (int kk = 0; kk < 4; ++kk) {
    v16h a = load_a_frag(&A[0][0], 128, lane, kk * 32);
    v16h b = load_b_packed(eW2p, 4, wv, kk, lane);
    c = __builtin_amdgcn_wmma_f32_16x16x32_f16(false, a, false, b, (short)0, c,
                                               false, false);
  }
  int n = n0 + (lane & 15);
  int mb = (lane & 16) ? 8 : 0;
  if (e0 + 16 <= E) {
#pragma unroll
    for (int r = 0; r < 8; ++r)
      hat_e[(size_t)(e0 + mb + r) * 128 + n] = c[r] + eb2[n];
  } else {
#pragma unroll
    for (int r = 0; r < 8; ++r) {
      int ge = e0 + mb + r;
      if (ge < E) hat_e[(size_t)ge * 128 + n] = c[r] + eb2[n];
    }
  }
}

// ---------------------------------------------------------------------------
// K=128 WMMA GEMM, 32-row tiles: out[M,128] = A[M,128] @ B (+bias)
// Each wave: one 16-col tile, two 16-row tiles sharing the B fragment.
// ---------------------------------------------------------------------------
template <bool HAS_BIAS>
__global__ void gemm128_kernel(const float* __restrict__ Am,
                               const _Float16* __restrict__ Bp,
                               const float* __restrict__ bias,
                               float* __restrict__ out, int M) {
  __shared__ _Float16 ldsA[32 * 128];
  int row0 = blockIdx.x * 32;
  int tid = threadIdx.x;
  bool full = (row0 + 32 <= M);
  if (full) {
    const float* src = Am + (size_t)row0 * 128;
    for (int i = tid; i < 32 * 128; i += 256) ldsA[i] = (_Float16)src[i];
  } else {
    for (int i = tid; i < 32 * 128; i += 256) {
      int gr = row0 + (i >> 7);
      ldsA[i] = (_Float16)((gr < M) ? Am[(size_t)gr * 128 + (i & 127)] : 0.0f);
    }
  }
  __syncthreads();
  int lane = tid & 31, wv = tid >> 5;
  int n0 = wv * 16;
  v8f c0 = {}, c1 = {};
#pragma unroll
  for (int kk = 0; kk < 4; ++kk) {
    v16h b = load_b_packed(Bp, 4, wv, kk, lane);
    v16h a0 = load_a_frag(ldsA, 128, lane, kk * 32);
    v16h a1 = load_a_frag(ldsA + 16 * 128, 128, lane, kk * 32);
    c0 = __builtin_amdgcn_wmma_f32_16x16x32_f16(false, a0, false, b, (short)0,
                                                c0, false, false);
    c1 = __builtin_amdgcn_wmma_f32_16x16x32_f16(false, a1, false, b, (short)0,
                                                c1, false, false);
  }
  int n = n0 + (lane & 15);
  int mb = (lane & 16) ? 8 : 0;
  float bv = HAS_BIAS ? bias[n] : 0.0f;
  if (full) {
#pragma unroll
    for (int r = 0; r < 8; ++r) {
      out[(size_t)(row0 + mb + r) * 128 + n] = c0[r] + bv;
      out[(size_t)(row0 + 16 + mb + r) * 128 + n] = c1[r] + bv;
    }
  } else {
#pragma unroll
    for (int r = 0; r < 8; ++r) {
      int g0 = row0 + mb + r;
      int g1 = g0 + 16;
      if (g0 < M) out[(size_t)g0 * 128 + n] = c0[r] + bv;
      if (g1 < M) out[(size_t)g1 * 128 + n] = c1[r] + bv;
    }
  }
}

// ---------------------------------------------------------------------------
// K=256 gather-concat WMMA GEMM, 32-row tiles:
//   row gr: A = cat(P[idxP ? idxP[gr] : gr], Q[idxQ ? idxQ[gr] : gr])
//   ACCUM:  out[gr] += A @ B + bias
//   !ACCUM: out[gr]  = act(A @ B + bias)   (RELU optional)
// ---------------------------------------------------------------------------
template <bool ACCUM, bool RELU>
__global__ void gemm256_gather_kernel(const float* __restrict__ P,
                                      const float* __restrict__ Q,
                                      const int* __restrict__ idxP,
                                      const int* __restrict__ idxQ,
                                      const _Float16* __restrict__ Bp,
                                      const float* __restrict__ bias,
                                      float* __restrict__ out, int M) {
  __shared__ _Float16 ldsA[32 * 256];
  int row0 = blockIdx.x * 32;
  int tid = threadIdx.x;
  bool full = (row0 + 32 <= M);
  for (int i = tid; i < 32 * 256; i += 256) {
    int r = i >> 8, c = i & 255;
    int gr = row0 + r;
    float v = 0.f;
    if (full || gr < M) {
      if (c < 128) {
        int s = idxP ? idxP[gr] : gr;
        v = P[(size_t)s * 128 + c];
      } else {
        int s = idxQ ? idxQ[gr] : gr;
        v = Q[(size_t)s * 128 + (c - 128)];
      }
    }
    ldsA[i] = (_Float16)v;
  }
  __syncthreads();
  int lane = tid & 31, wv = tid >> 5;
  int n0 = wv * 16;
  v8f c0 = {}, c1 = {};
#pragma unroll
  for (int kk = 0; kk < 8; ++kk) {
    v16h b = load_b_packed(Bp, 8, wv, kk, lane);
    v16h a0 = load_a_frag(ldsA, 256, lane, kk * 32);
    v16h a1 = load_a_frag(ldsA + 16 * 256, 256, lane, kk * 32);
    c0 = __builtin_amdgcn_wmma_f32_16x16x32_f16(false, a0, false, b, (short)0,
                                                c0, false, false);
    c1 = __builtin_amdgcn_wmma_f32_16x16x32_f16(false, a1, false, b, (short)0,
                                                c1, false, false);
  }
  int n = n0 + (lane & 15);
  int mb = (lane & 16) ? 8 : 0;
  float bv = bias[n];
  if (full) {
#pragma unroll
    for (int r = 0; r < 8; ++r) {
      size_t o0 = (size_t)(row0 + mb + r) * 128 + n;
      size_t o1 = (size_t)(row0 + 16 + mb + r) * 128 + n;
      if (ACCUM) {
        out[o0] += c0[r] + bv;
        out[o1] += c1[r] + bv;
      } else {
        float v0 = c0[r] + bv, v1 = c1[r] + bv;
        if (RELU) { v0 = fmaxf(v0, 0.f); v1 = fmaxf(v1, 0.f); }
        out[o0] = v0;
        out[o1] = v1;
      }
    }
  } else {
#pragma unroll
    for (int r = 0; r < 8; ++r) {
      int g0 = row0 + mb + r;
      int g1 = g0 + 16;
      if (g0 < M) {
        float v = c0[r] + bv;
        if (ACCUM) out[(size_t)g0 * 128 + n] += v;
        else {
          if (RELU) v = fmaxf(v, 0.f);
          out[(size_t)g0 * 128 + n] = v;
        }
      }
      if (g1 < M) {
        float v = c1[r] + bv;
        if (ACCUM) out[(size_t)g1 * 128 + n] += v;
        else {
          if (RELU) v = fmaxf(v, 0.f);
          out[(size_t)g1 * 128 + n] = v;
        }
      }
    }
  }
}

// ---------------------------------------------------------------------------
// GCN helpers
// ---------------------------------------------------------------------------
__global__ void edge_mean_kernel(const float* __restrict__ hat_e,
                                 float* __restrict__ w, int E) {
  int e = blockIdx.x * blockDim.x + threadIdx.x;
  if (e >= E) return;
  const float4* p = (const float4*)(hat_e + (size_t)e * 128);
  float s = 0.f;
#pragma unroll 8
  for (int i = 0; i < 32; ++i) {
    float4 v = p[i];
    s += v.x + v.y + v.z + v.w;
  }
  w[e] = s * (1.0f / 128.0f);
}

__global__ void deg_accum_kernel(const int* __restrict__ dst,
                                 const float* __restrict__ w,
                                 float* __restrict__ deg, int E) {
  int e = blockIdx.x * blockDim.x + threadIdx.x;
  if (e >= E) return;
  atomicAdd(&deg[dst[e]], w ? w[e] : 1.0f);
}

__global__ void dis_kernel(float* __restrict__ deg, int N) {
  int i = blockIdx.x * blockDim.x + threadIdx.x;
  if (i < N) deg[i] = rsqrtf(fmaxf(deg[i], 1e-6f));
}

// agg[dst] += dis[src]*w*dis[dst] * xw[src] ; thread handles 4 columns
__global__ void scatter_kernel(const float* __restrict__ xw,
                               const int* __restrict__ src,
                               const int* __restrict__ dst,
                               const float* __restrict__ w,
                               const float* __restrict__ dis,
                               float* __restrict__ agg, int E) {
  int idx = blockIdx.x * blockDim.x + threadIdx.x;
  int e = idx >> 5;
  if (e >= E) return;
  int q = idx & 31;
  int s = src[e], d = dst[e];
  float nw = dis[s] * (w ? w[e] : 1.0f) * dis[d];
  const float4 v = *(const float4*)(xw + (size_t)s * 128 + q * 4);
  float* a = agg + (size_t)d * 128 + q * 4;
  atomicAdd(a + 0, nw * v.x);
  atomicAdd(a + 1, nw * v.y);
  atomicAdd(a + 2, nw * v.z);
  atomicAdd(a + 3, nw * v.w);
}

// out += relu(agg + dis^2 * xw + bias) (+ m if non-null)
__global__ void node_update_kernel(float* __restrict__ out,
                                   const float* __restrict__ agg,
                                   const float* __restrict__ xw,
                                   const float* __restrict__ dis,
                                   const float* __restrict__ bias,
                                   const float* __restrict__ m, int N) {
  int idx = blockIdx.x * blockDim.x + threadIdx.x;
  if (idx >= N * 128) return;
  int n = idx >> 7, c = idx & 127;
  float di = dis[n];
  float v = fmaxf(agg[idx] + di * di * xw[idx] + bias[c], 0.0f);
  out[idx] += v + (m ? m[idx] : 0.0f);
}

// ---------------------------------------------------------------------------
// Output heads
// ---------------------------------------------------------------------------
__global__ void heads_kernel(const float* __restrict__ hat_h,
                             const float* __restrict__ h,
                             const float* __restrict__ aW,
                             const float* __restrict__ ab,
                             const float* __restrict__ baW,
                             const float* __restrict__ bab,
                             const float* __restrict__ cW,
                             const float* __restrict__ cbv,
                             const float* __restrict__ xW,
                             const float* __restrict__ xbv,
                             const float* __restrict__ old_chi,
                             const float* __restrict__ old_phi,
                             const float* __restrict__ old_psi,
                             const float* __restrict__ old_cb,
                             const float* __restrict__ old_ca,
                             float* __restrict__ out, int N) {
  int n = blockIdx.x * blockDim.x + threadIdx.x;
  if (n >= N) return;
  float da = 0.f, c0 = 0.f, c1 = 0.f, c2 = 0.f;
  for (int r = 0; r < 128; ++r) {
    float s = fmaxf(hat_h[(size_t)n * 128 + r], 0.0f);
    da += s * aW[r];
    c0 += s * cW[r * 3 + 0];
    c1 += s * cW[r * 3 + 1];
    c2 += s * cW[r * 3 + 2];
  }
  float db = 0.f, x0 = 0.f, x1 = 0.f, x2 = 0.f;
  for (int r = 0; r < 128; ++r) {
    float t = fmaxf(h[(size_t)n * 128 + r], 0.0f);
    db += t * baW[r];
    x0 += t * xW[r * 3 + 0];
    x1 += t * xW[r * 3 + 1];
    x2 += t * xW[r * 3 + 2];
  }
  float dchi = da + ab[0];
  float dphi = db + bab[0];
  float* chi = out;
  float* phi = out + 4 * N;
  float* psi = out + 5 * N;
  float* cbp = out + 6 * N;
  float* cap = out + 9 * N;
#pragma unroll
  for (int k = 0; k < 4; ++k)
    chi[k * N + n] = wrapf(old_chi[k * N + n] + dchi);
  phi[n] = wrapf(old_phi[n] + dphi);
  psi[n] = wrapf(old_psi[n] + dphi);
  cbp[n * 3 + 0] = old_cb[n * 3 + 0] + c0 + cbv[0];
  cbp[n * 3 + 1] = old_cb[n * 3 + 1] + c1 + cbv[1];
  cbp[n * 3 + 2] = old_cb[n * 3 + 2] + c2 + cbv[2];
  cap[n * 3 + 0] = old_ca[n * 3 + 0] + x0 + xbv[0];
  cap[n * 3 + 1] = old_ca[n * 3 + 1] + x1 + xbv[1];
  cap[n * 3 + 2] = old_ca[n * 3 + 2] + x2 + xbv[2];
}

// ---------------------------------------------------------------------------
// Host orchestration
// ---------------------------------------------------------------------------
extern "C" void kernel_launch(void* const* d_in, const int* in_sizes, int n_in,
                              void* d_out, int out_size, void* d_ws,
                              size_t ws_size, hipStream_t stream) {
  (void)n_in; (void)out_size; (void)ws_size;
  const float* V_com   = (const float*)d_in[0];
  const float* SBF_chi = (const float*)d_in[1];
  const float* X_ca    = (const float*)d_in[2];
  const float* V_cb    = (const float*)d_in[3];
  const float* SBF_phi = (const float*)d_in[4];
  const float* SBF_psi = (const float*)d_in[5];
  const float* d_sc    = (const float*)d_in[6];
  const float* d_min   = (const float*)d_in[7];
  const float* old_chi = (const float*)d_in[8];
  const float* old_phi = (const float*)d_in[9];
  const float* old_psi = (const float*)d_in[10];
  const float* old_cb  = (const float*)d_in[11];
  const float* old_ca  = (const float*)d_in[12];
  const float* res_tab = (const float*)d_in[13];
  const float* sc_ln_g = (const float*)d_in[14];
  const float* sc_ln_b = (const float*)d_in[15];
  const float* sc_W    = (const float*)d_in[16];
  const float* sc_b    = (const float*)d_in[17];
  const float* bb_ln_g = (const float*)d_in[18];
  const float* bb_ln_b = (const float*)d_in[19];
  const float* bb_W    = (const float*)d_in[20];
  const float* bb_b    = (const float*)d_in[21];
  const float* e_ln_g  = (const float*)d_in[22];
  const float* e_ln_b  = (const float*)d_in[23];
  const float* eW1     = (const float*)d_in[24];
  const float* eb1     = (const float*)d_in[25];
  const float* eW2     = (const float*)d_in[26];
  const float* eb2     = (const float*)d_in[27];
  const float* euW     = (const float*)d_in[28];
  const float* eub     = (const float*)d_in[29];
  const float* gW      = (const float*)d_in[30];
  const float* gb      = (const float*)d_in[31];
  const float* bgW     = (const float*)d_in[32];
  const float* bgb     = (const float*)d_in[33];
  const float* iW1     = (const float*)d_in[34];
  const float* ib1     = (const float*)d_in[35];
  const float* iW2     = (const float*)d_in[36];
  const float* ib2     = (const float*)d_in[37];
  const float* aW      = (const float*)d_in[38];
  const float* ab      = (const float*)d_in[39];
  const float* baW     = (const float*)d_in[40];
  const float* bab     = (const float*)d_in[41];
  const float* cW      = (const float*)d_in[42];
  const float* cb      = (const float*)d_in[43];
  const float* xW      = (const float*)d_in[44];
  const float* xb      = (const float*)d_in[45];
  const int* ridx_i    = (const int*)d_in[46];
  const int* ridx_j    = (const int*)d_in[47];
  const int* esc       = (const int*)d_in[48];
  const int* ebb       = (const int*)d_in[49];

  const int N  = in_sizes[0] / 3;
  const int E  = in_sizes[6];
  const int EB = in_sizes[49] / 2;
  const int* src  = esc;
  const int* dst  = esc + E;
  const int* bsrc = ebb;
  const int* bdst = ebb + EB;

  // ---- workspace carve-out ----
  char* cur = (char*)d_ws;
  auto alloc = [&](size_t bytes) -> char* {
    char* p = cur;
    cur += (bytes + 255) & ~(size_t)255;
    return p;
  };
  float* hat_h = (float*)alloc((size_t)N * 128 * 4);
  float* h_buf = (float*)alloc((size_t)N * 128 * 4);
  float* hat_e = (float*)alloc((size_t)E * 128 * 4);
  float* bufA  = (float*)alloc((size_t)N * 128 * 4);  // xw / interaction hidden
  float* bufB  = (float*)alloc((size_t)N * 128 * 4);  // agg
  float* bufC  = (float*)alloc((size_t)N * 128 * 4);  // m
  float* ew    = (float*)alloc((size_t)E * 4);        // edge weights
  float* dis   = (float*)alloc((size_t)N * 4);        // deg -> dis in place
  _Float16* eW2p = (_Float16*)alloc(128 * 128 * 2);
  _Float16* euWp = (_Float16*)alloc(256 * 128 * 2);
  _Float16* gWp  = (_Float16*)alloc(128 * 128 * 2);
  _Float16* bgWp = (_Float16*)alloc(128 * 128 * 2);
  _Float16* iW1p = (_Float16*)alloc(256 * 128 * 2);
  _Float16* iW2p = (_Float16*)alloc(128 * 128 * 2);

  auto cdiv = [](int a, int b) { return (a + b - 1) / b; };

  // ---- weight packing into fragment-major f16 ----
  pack_b_kernel<<<cdiv(128 * 128, 256), 256, 0, stream>>>(eW2, eW2p, 128);
  pack_b_kernel<<<cdiv(256 * 128, 256), 256, 0, stream>>>(euW, euWp, 256);
  pack_b_kernel<<<cdiv(128 * 128, 256), 256, 0, stream>>>(gW, gWp, 128);
  pack_b_kernel<<<cdiv(128 * 128, 256), 256, 0, stream>>>(bgW, bgWp, 128);
  pack_b_kernel<<<cdiv(256 * 128, 256), 256, 0, stream>>>(iW1, iW1p, 256);
  pack_b_kernel<<<cdiv(128 * 128, 256), 256, 0, stream>>>(iW2, iW2p, 128);

  // ---- node featurization ----
  node_init_kernel<<<N, 128, 0, stream>>>(V_com, SBF_chi, X_ca, V_cb, SBF_phi,
                                          SBF_psi, sc_ln_g, sc_ln_b, sc_W, sc_b,
                                          bb_ln_g, bb_ln_b, bb_W, bb_b, hat_h,
                                          h_buf, N);

  // ---- edge MLP ----
  edge_mlp_kernel<<<cdiv(E, 16), 256, 0, stream>>>(d_sc, d_min, res_tab, ridx_i,
                                                   ridx_j, e_ln_g, e_ln_b, eW1,
                                                   eb1, eW2p, eb2, hat_e, E);

  // ---- 3 sidechain GCN layers ----
  for (int layer = 0; layer < 3; ++layer) {
    edge_mean_kernel<<<cdiv(E, 256), 256, 0, stream>>>(hat_e, ew, E);
    fill_kernel<<<cdiv(N, 256), 256, 0, stream>>>(dis, 1.0f, N);
    deg_accum_kernel<<<cdiv(E, 256), 256, 0, stream>>>(dst, ew, dis, E);
    dis_kernel<<<cdiv(N, 256), 256, 0, stream>>>(dis, N);
    gemm128_kernel<false><<<cdiv(N, 32), 256, 0, stream>>>(hat_h, gWp, nullptr,
                                                           bufA, N);  // xw
    fill_kernel<<<cdiv(N * 128, 256), 256, 0, stream>>>(bufB, 0.0f, N * 128);
    scatter_kernel<<<cdiv(E * 32, 256), 256, 0, stream>>>(bufA, src, dst, ew,
                                                          dis, bufB, E);
    node_update_kernel<<<cdiv(N * 128, 256), 256, 0, stream>>>(
        hat_h, bufB, bufA, dis, gb, nullptr, N);
    // hat_e += cat(hat_h[src], hat_h[dst]) @ euW + eub
    gemm256_gather_kernel<true, false><<<cdiv(E, 32), 256, 0, stream>>>(
        hat_h, hat_h, src, dst, euWp, eub, hat_e, E);
  }

  // ---- 2 backbone layers ----
  for (int layer = 0; layer < 2; ++layer) {
    // m = relu(cat(h, hat_h) @ iW1 + ib1) @ iW2 + ib2
    gemm256_gather_kernel<false, true><<<cdiv(N, 32), 256, 0, stream>>>(
        h_buf, hat_h, nullptr, nullptr, iW1p, ib1, bufA, N);
    gemm128_kernel<true><<<cdiv(N, 32), 256, 0, stream>>>(bufA, iW2p, ib2,
                                                          bufC, N);
    // gcn on h with unit edge weights
    fill_kernel<<<cdiv(N, 256), 256, 0, stream>>>(dis, 1.0f, N);
    deg_accum_kernel<<<cdiv(EB, 256), 256, 0, stream>>>(bdst, nullptr, dis, EB);
    dis_kernel<<<cdiv(N, 256), 256, 0, stream>>>(dis, N);
    gemm128_kernel<false><<<cdiv(N, 32), 256, 0, stream>>>(h_buf, bgWp, nullptr,
                                                           bufA, N);  // xw
    fill_kernel<<<cdiv(N * 128, 256), 256, 0, stream>>>(bufB, 0.0f, N * 128);
    scatter_kernel<<<cdiv(EB * 32, 256), 256, 0, stream>>>(bufA, bsrc, bdst,
                                                           nullptr, dis, bufB,
                                                           EB);
    node_update_kernel<<<cdiv(N * 128, 256), 256, 0, stream>>>(
        h_buf, bufB, bufA, dis, bgb, bufC, N);
  }

  // ---- heads ----
  heads_kernel<<<cdiv(N, 256), 256, 0, stream>>>(
      hat_h, h_buf, aW, ab, baW, bab, cW, cb, xW, xb, old_chi, old_phi, old_psi,
      old_cb, old_ca, (float*)d_out, N);
}
